// Model_63995012710934
// MI455X (gfx1250) — compile-verified, matrix operand order checked by reading, not run on hardware
//
#include <hip/hip_runtime.h>
#include <hip/hip_bf16.h>

// ---------------------------------------------------------------------------
// Axial cross-attention (C=256, HEADS=8, W=251) for gfx1250 / MI455X.
// 5-kernel fused pipeline:
//   - all GEMM-shaped math on v_wmma_f32_16x16x32_f16 (wave32)
//   - B operands stored fragment-major in LDS -> ds_load_b128 (no u16 storms)
//   - raw projections kept in f16; staged into LDS via TENSOR_LOAD_TO_LDS
//     (Tensor Data Mover) with s_wait_tensorcnt, overlapped with compute
// ---------------------------------------------------------------------------

typedef __attribute__((ext_vector_type(16))) _Float16 v16h;
typedef __attribute__((ext_vector_type(8)))  float    v8f;
typedef unsigned int u32x4 __attribute__((ext_vector_type(4)));
typedef int          i32x8 __attribute__((ext_vector_type(8)));
typedef int          i32x4 __attribute__((ext_vector_type(4)));

#define NBB   30      // B*H axial batches
#define WLEN  251
#define WPAD  256
#define EPSB  1e-5f

#ifndef __has_builtin
#define __has_builtin(x) 0
#endif
#if __has_builtin(__builtin_amdgcn_tensor_load_to_lds) && __has_builtin(__builtin_amdgcn_s_wait_tensorcnt)
#define USE_TDM 1
#else
#define USE_TDM 0
#endif

// ---- TDM: DMA `rows` x 256 f16 row-block (row stride 256) global -> LDS ----
#if USE_TDM
__device__ __forceinline__ void tdm_load_rows_f16(unsigned lds_off, const _Float16* gsrc, int rows) {
  unsigned long long ga = (unsigned long long)(uintptr_t)gsrc;
  u32x4 g0 = { 1u,                                        // count=1
               lds_off,                                   // lds_addr
               (unsigned)(ga & 0xFFFFFFFFull),            // global_addr lo
               (unsigned)((ga >> 32) & 0x01FFFFFFull) | (2u << 30) };  // addr hi | type=2
  i32x8 g1 = { (int)(1u << 16),                 // data_size=1 (2 bytes)
               (int)((unsigned)WPAD << 16),     // tensor_dim0 = 256
               (int)((unsigned)rows << 16),     // tensor_dim1 = rows
               (int)((unsigned)WPAD << 16),     // tile_dim0 = 256
               rows,                            // tile_dim1 = rows, tile_dim2=0
               WPAD,                            // tensor_dim0_stride lo = 256
               0, 0 };
  i32x4 gz4 = { 0, 0, 0, 0 };
  i32x8 gz8 = { 0, 0, 0, 0, 0, 0, 0, 0 };
  __builtin_amdgcn_tensor_load_to_lds(g0, g1, gz4, gz4, gz8, 0);
}
#endif

// ---- WMMA fragment helpers (wave32, 16-bit 16x16x32) ----------------------
// A (MxK=16x32) from row-major [m][ld]: per-lane reads are 2 contiguous runs.
__device__ __forceinline__ v16h load_fragA(const _Float16* __restrict__ src, int ld, int lane) {
  v16h a;
  const int m = lane & 15;
  const int kbase = (lane >> 4) * 8;
#pragma unroll
  for (int j = 0; j < 16; ++j) {
    int k = kbase + (j & 7) + ((j >> 3) << 4);
    a[j] = src[m * ld + k];
  }
  return a;
}
// B fragment from fragment-major storage: 16 contiguous halves per lane.
__device__ __forceinline__ v16h load_fragP(const _Float16* __restrict__ src) {
  v16h r;
#pragma unroll
  for (int j = 0; j < 16; ++j) r[j] = src[j];
  return r;
}
// swizzle for B fragment-major store: element (k_local 0..31, n_local 0..15)
__device__ __forceinline__ int bfrag_idx(int kl, int nl) {
  int lane = nl | (((kl >> 3) & 1) << 4);
  int j = ((kl >> 4) & 1) * 8 + (kl & 7);
  return lane * 16 + j;
}

// ---------------------------------------------------------------------------
// K1: projections -> Pf (f16). P[bb][o][w]; o: 0..127 wk@x, 128..255 wq@y,
// 256..511 wv@x.  grid (8 w-blocks of 32, 8 o-blocks of 64, 30), 256 thr.
// ---------------------------------------------------------------------------
__global__ __launch_bounds__(256)
void k_proj(const float* __restrict__ x, const float* __restrict__ y,
            const float* __restrict__ wk, const float* __restrict__ wq,
            const float* __restrict__ wv, _Float16* __restrict__ Pf) {
  const int wb = blockIdx.x, ob = blockIdx.y, bb = blockIdx.z;
  const int o0 = ob * 64, w0 = wb * 32;
  const int b = bb / 15, hh = bb % 15;
  __shared__ _Float16 Wt[64][264];   // A: weights [o][c]
  __shared__ _Float16 XB[8192];      // B fragment-major: [kc][nt][lane][16]
  const int tid = threadIdx.x;

  const float* wsrc; int obase;
  if (o0 < 128)      { wsrc = wk; obase = o0;       }
  else if (o0 < 256) { wsrc = wq; obase = o0 - 128; }
  else               { wsrc = wv; obase = o0 - 256; }
  for (int i = tid; i < 64 * 256; i += 256) {
    int r = i >> 8, c = i & 255;
    Wt[r][c] = (_Float16)wsrc[(obase + r) * 256 + c];
  }
  const float* src = (o0 >= 128 && o0 < 256) ? y : x;
  for (int i = tid; i < 256 * 32; i += 256) {
    int c = i >> 5, wi = i & 31;
    int w = w0 + wi;
    float v = (w < WLEN) ? src[((b * 256 + c) * 15 + hh) * WLEN + w] : 0.f;
    XB[((c >> 5) * 2 + (wi >> 4)) * 512 + bfrag_idx(c & 31, wi & 15)] = (_Float16)v;
  }
  __syncthreads();

  const int lane = tid & 31, wid = tid >> 5;
  const int mo = (wid >> 1) * 16, nt = wid & 1;
  v8f acc = {};
#pragma unroll
  for (int kc = 0; kc < 8; ++kc) {
    v16h a  = load_fragA(&Wt[mo][kc * 32], 264, lane);
    v16h bf = load_fragP(&XB[(kc * 2 + nt) * 512 + lane * 16]);
    acc = __builtin_amdgcn_wmma_f32_16x16x32_f16(false, a, false, bf, (short)0, acc, false, false);
  }
  const int nloc = lane & 15, mh = (lane >> 4) * 8;
#pragma unroll
  for (int vi = 0; vi < 8; ++vi) {
    int o = o0 + mo + mh + vi;
    int w = w0 + nt * 16 + nloc;
    Pf[(bb * 512 + o) * WPAD + w] = (_Float16)acc[vi];
  }
}

// ---------------------------------------------------------------------------
// K2: per-channel BN stats over P -> fused scale/bias.  grid(512), 256 thr.
// ---------------------------------------------------------------------------
__global__ __launch_bounds__(256)
void k_projstats(const _Float16* __restrict__ Pf,
                 const float* __restrict__ gk,  const float* __restrict__ bk,
                 const float* __restrict__ gq,  const float* __restrict__ bq,
                 const float* __restrict__ gkq, const float* __restrict__ bkq,
                 float* __restrict__ projSB) {
  const int o = blockIdx.x;
  __shared__ float rs[256], rs2[256];
  float s = 0.f, s2 = 0.f;
  for (int i = threadIdx.x; i < NBB * WLEN; i += 256) {
    int bb = i / WLEN, w = i % WLEN;
    float v = (float)Pf[(bb * 512 + o) * WPAD + w];
    s += v; s2 += v * v;
  }
  rs[threadIdx.x] = s; rs2[threadIdx.x] = s2;
  __syncthreads();
  for (int st = 128; st > 0; st >>= 1) {
    if (threadIdx.x < st) {
      rs[threadIdx.x]  += rs[threadIdx.x + st];
      rs2[threadIdx.x] += rs2[threadIdx.x + st];
    }
    __syncthreads();
  }
  if (threadIdx.x == 0) {
    const float n = (float)(NBB * WLEN);
    float m = rs[0] / n;
    float var = rs2[0] / n - m * m;
    float g, bia;
    if (o < 128)      { g = gk[o];        bia = bk[o];        }
    else if (o < 256) { g = gq[o - 128];  bia = bq[o - 128];  }
    else              { g = gkq[o - 256]; bia = bkq[o - 256]; }
    float sc = g * rsqrtf(var + EPSB);
    projSB[o * 2 + 0] = sc;
    projSB[o * 2 + 1] = bia - m * sc;
  }
}

// ---------------------------------------------------------------------------
// K3: logit BN statistics (qk/qr/kr sum,sumsq) per (bb,h,xtile).
// grid (16, 8, 30), 256 thr.  TDM-staged k/q; WMMA qk; VALU Toeplitz.
// ---------------------------------------------------------------------------
__global__ __launch_bounds__(256)
void k_logitstats(const _Float16* __restrict__ Pf, const float* __restrict__ projSB,
                  const float* __restrict__ rel, float* __restrict__ lpart) {
  const int xt = blockIdx.x, h = blockIdx.y, bb = blockIdx.z;
  const int x0 = xt * 16;
  __shared__ _Float16 Praw[32][256];   // raw k(0..15)/q(16..31) rows, pre-BN
  __shared__ _Float16 khB[8192];       // k as B fragments: [yt][lane][16]
  __shared__ _Float16 kf16[16][264];   // plain normalized k for kr VALU
  __shared__ _Float16 qA[16][32];
  __shared__ float    qf[16][17];
  __shared__ float    red[6][264];
  const int tid = threadIdx.x;
  const int lane = tid & 31, wid = tid >> 5;

  const _Float16* gkq_rows = Pf + (size_t)(bb * 512 + h * 64) * WPAD;
#if USE_TDM
  if (wid == 0) tdm_load_rows_f16((unsigned)(uintptr_t)(void*)&Praw[0][0], gkq_rows, 32);
  __builtin_amdgcn_s_wait_tensorcnt((short)0);
#else
  for (int i = tid; i < 32 * 256; i += 256) Praw[i >> 8][i & 255] = gkq_rows[i];
#endif
  __syncthreads();

  for (int i = tid; i < 32 * 256; i += 256) {          // build khB (+ kf16)
    int d = i >> 8, yy = i & 255;
    float v = 0.f;
    if (d < 16 && yy < WLEN) {
      int o = h * 64 + d;
      v = (float)Praw[d][yy] * projSB[o * 2] + projSB[o * 2 + 1];
    }
    khB[(yy >> 4) * 512 + bfrag_idx(d, yy & 15)] = (_Float16)v;
    if (d < 16) kf16[d][yy] = (_Float16)v;
  }
  for (int i = tid; i < 256; i += 256) {               // q tile [x][d], K-padded
    int d = i >> 4, xi = i & 15;
    int o = h * 64 + 16 + d;
    int xx = x0 + xi;
    float v = (xx < WLEN) ? (float)Praw[16 + d][xx] * projSB[o * 2] + projSB[o * 2 + 1] : 0.f;
    qf[d][xi] = v;
    qA[xi][d] = (_Float16)v;
    qA[xi][d + 16] = (_Float16)0.f;
  }
  __syncthreads();

  float s[6] = {0.f, 0.f, 0.f, 0.f, 0.f, 0.f};
  v16h a = load_fragA(&qA[0][0], 32, lane);
#pragma unroll
  for (int t = 0; t < 2; ++t) {
    int yt = wid * 2 + t;
    v16h bf = load_fragP(&khB[yt * 512 + lane * 16]);
    v8f acc = {};
    acc = __builtin_amdgcn_wmma_f32_16x16x32_f16(false, a, false, bf, (short)0, acc, false, false);
    const int nloc = lane & 15, mh = (lane >> 4) * 8;
#pragma unroll
    for (int vi = 0; vi < 8; ++vi) {
      int xi = mh + vi;
      int xx = x0 + xi;
      int yv = yt * 16 + nloc;
      if (xx < WLEN && yv < WLEN) {
        float qk = acc[vi];
        float qr = 0.f, kr = 0.f;
        const int dqy = yv - xx + 250;
        const int dky = xx - yv + 250;
#pragma unroll
        for (int d = 0; d < 16; ++d) {
          qr += qf[d][xi] * rel[d * 501 + dqy];
          kr += (float)kf16[d][yv] * rel[(16 + d) * 501 + dky];
        }
        s[0] += qk; s[1] += qk * qk;
        s[2] += qr; s[3] += qr * qr;
        s[4] += kr; s[5] += kr * kr;
      }
    }
  }
#pragma unroll
  for (int t = 0; t < 6; ++t) red[t][tid] = s[t];
  __syncthreads();
  for (int st = 128; st > 0; st >>= 1) {
    if (tid < st) {
#pragma unroll
      for (int t = 0; t < 6; ++t) red[t][tid] += red[t][tid + st];
    }
    __syncthreads();
  }
  if (tid < 6) {
    int wg = (bb * 8 + h) * 16 + xt;
    lpart[wg * 6 + tid] = red[tid][0];
  }
}

// ---------------------------------------------------------------------------
// K4: finalize logit BN: scales sstats[0..23], per-head offsets sstats[24..31].
// ---------------------------------------------------------------------------
__global__ __launch_bounds__(256)
void k_logitfin(const float* __restrict__ lpart, const float* __restrict__ glog,
                const float* __restrict__ blog, float* __restrict__ sstats) {
  __shared__ float msh[24], ssh[24];
  const int tid = threadIdx.x;
  if (tid < 24) {
    const int t = tid >> 3, h = tid & 7;               // channel = t*8+h
    float s = 0.f, s2 = 0.f;
    for (int bb = 0; bb < NBB; ++bb)
      for (int xt = 0; xt < 16; ++xt) {
        int wg = (bb * 8 + h) * 16 + xt;
        s  += lpart[wg * 6 + t * 2];
        s2 += lpart[wg * 6 + t * 2 + 1];
      }
    const float n = (float)NBB * (float)WLEN * (float)WLEN;
    float m = s / n, var = s2 / n - m * m;
    float sc = glog[tid] * rsqrtf(var + EPSB);
    msh[tid] = m; ssh[tid] = sc;
    sstats[tid] = sc;
  }
  __syncthreads();
  if (tid < 8) {
    float off = 0.f;
    for (int t = 0; t < 3; ++t) off += blog[t * 8 + tid] - msh[t * 8 + tid] * ssh[t * 8 + tid];
    sstats[24 + tid] = off;
  }
}

// ---------------------------------------------------------------------------
// K5: recompute logits, softmax, attention (W@v^T WMMA), attn_enc, output.
// grid (16, 8, 30), 256 thr.  TDM of v-rows overlapped with logits compute.
// ---------------------------------------------------------------------------
__global__ __launch_bounds__(256)
void k_attn(const _Float16* __restrict__ Pf, const float* __restrict__ projSB,
            const float* __restrict__ rel, const float* __restrict__ sstats,
            float* __restrict__ out) {
  const int xt = blockIdx.x, h = blockIdx.y, bb = blockIdx.z;
  const int x0 = xt * 16;
  const int b = bb / 15, hh = bb % 15;

  __shared__ _Float16 Praw[32][256];   // TDM target: k/q rows, later v rows
  __shared__ _Float16 un[8192];        // khB (phase 1) / vB (phase 2) fragments
  __shared__ _Float16 un2[4224];       // kf16 (phase 1) / Wh (phase 3)
  __shared__ float    Ls[16][260];
  __shared__ _Float16 qA[16][32];
  __shared__ float    qf[16][17];
  __shared__ float    Aout[32][17];
  __shared__ float    rred[16][17];
  __shared__ float    rowstat[16];
  _Float16* khB = un;
  _Float16* vB  = un;
  _Float16 (*kf16)[264] = (_Float16(*)[264])un2;
  _Float16 (*Wh)[260]   = (_Float16(*)[260])un2;

  const int tid = threadIdx.x;
  const int lane = tid & 31, wid = tid >> 5;
  const float s0 = sstats[h], s1 = sstats[8 + h], s2 = sstats[16 + h], soff = sstats[24 + h];

  // ---- stage k/q rows ----
  const _Float16* gkq_rows = Pf + (size_t)(bb * 512 + h * 64) * WPAD;
  const _Float16* gv_rows  = gkq_rows + 32 * WPAD;
#if USE_TDM
  if (wid == 0) tdm_load_rows_f16((unsigned)(uintptr_t)(void*)&Praw[0][0], gkq_rows, 32);
  __builtin_amdgcn_s_wait_tensorcnt((short)0);
#else
  for (int i = tid; i < 32 * 256; i += 256) Praw[i >> 8][i & 255] = gkq_rows[i];
#endif
  __syncthreads();

  // ---- build k/q structures (BN applied) ----
  for (int i = tid; i < 32 * 256; i += 256) {
    int d = i >> 8, yy = i & 255;
    float v = 0.f;
    if (d < 16 && yy < WLEN) {
      int o = h * 64 + d;
      v = (float)Praw[d][yy] * projSB[o * 2] + projSB[o * 2 + 1];
    }
    khB[(yy >> 4) * 512 + bfrag_idx(d, yy & 15)] = (_Float16)v;
    if (d < 16) kf16[d][yy] = (_Float16)v;
  }
  for (int i = tid; i < 256; i += 256) {
    int d = i >> 4, xi = i & 15;
    int o = h * 64 + 16 + d;
    int xx = x0 + xi;
    float v = (xx < WLEN) ? (float)Praw[16 + d][xx] * projSB[o * 2] + projSB[o * 2 + 1] : 0.f;
    qf[d][xi] = v;
    qA[xi][d] = (_Float16)v;
    qA[xi][d + 16] = (_Float16)0.f;
  }
  __syncthreads();

  // ---- kick off async DMA of v rows (overlaps logits compute) ----
#if USE_TDM
  if (wid == 0) tdm_load_rows_f16((unsigned)(uintptr_t)(void*)&Praw[0][0], gv_rows, 32);
#else
  for (int i = tid; i < 32 * 256; i += 256) Praw[i >> 8][i & 255] = gv_rows[i];
#endif

  // ---- logits: s0*qk + s1*qr + s2*kr + soff ----
  {
    v16h a = load_fragA(&qA[0][0], 32, lane);
#pragma unroll
    for (int t = 0; t < 2; ++t) {
      int yt = wid * 2 + t;
      v16h bf = load_fragP(&khB[yt * 512 + lane * 16]);
      v8f acc = {};
      acc = __builtin_amdgcn_wmma_f32_16x16x32_f16(false, a, false, bf, (short)0, acc, false, false);
      const int nloc = lane & 15, mh = (lane >> 4) * 8;
#pragma unroll
      for (int vi = 0; vi < 8; ++vi) {
        int xi = mh + vi;
        int xx = x0 + xi;
        int yv = yt * 16 + nloc;
        float val = 0.f;
        if (xx < WLEN && yv < WLEN) {
          float qr = 0.f, kr = 0.f;
          const int dqy = yv - xx + 250;
          const int dky = xx - yv + 250;
#pragma unroll
          for (int d = 0; d < 16; ++d) {
            qr += qf[d][xi] * rel[d * 501 + dqy];
            kr += (float)kf16[d][yv] * rel[(16 + d) * 501 + dky];
          }
          val = s0 * acc[vi] + s1 * qr + s2 * kr + soff;
        }
        Ls[xi][yt * 16 + nloc] = val;
      }
    }
  }
  __syncthreads();
#if USE_TDM
  __builtin_amdgcn_s_wait_tensorcnt((short)0);
#endif
  __syncthreads();

  // ---- build v B-fragments (overwrites khB region) ----
  for (int i = tid; i < 32 * 256; i += 256) {
    int yy = i >> 5, d = i & 31;
    int o = h * 64 + 32 + d;
    float v = (yy < WLEN) ? (float)Praw[d][yy] * projSB[o * 2] + projSB[o * 2 + 1] : 0.f;
    vB[((yy >> 5) * 2 + (d >> 4)) * 512 + bfrag_idx(yy & 31, d & 15)] = (_Float16)v;
  }
  __syncthreads();

  // ---- softmax over y (16 threads per row); Wh overwrites kf16 region ----
  {
    const int row = tid >> 4, c = tid & 15;
    float mx = -1e30f;
    for (int j = 0; j < 16; ++j) {
      int yv = c + j * 16;
      if (yv < WLEN) mx = fmaxf(mx, Ls[row][yv]);
    }
    rred[row][c] = mx;
    __syncthreads();
    if (c == 0) {
      float m = rred[row][0];
      for (int j = 1; j < 16; ++j) m = fmaxf(m, rred[row][j]);
      rowstat[row] = m;
    }
    __syncthreads();
    const float m = rowstat[row];
    float sm = 0.f;
    for (int j = 0; j < 16; ++j) {
      int yv = c + j * 16;
      if (yv < WLEN) {
        float e = __expf(Ls[row][yv] - m);
        Ls[row][yv] = e;
        sm += e;
      }
    }
    rred[row][c] = sm;
    __syncthreads();
    if (c == 0) {
      float ssum = 0.f;
      for (int j = 0; j < 16; ++j) ssum += rred[row][j];
      rowstat[row] = 1.f / ssum;
    }
    __syncthreads();
    const float inv = rowstat[row];
    for (int j = 0; j < 16; ++j) {
      int yv = c + j * 16;
      float wv = (yv < WLEN) ? Ls[row][yv] * inv : 0.f;
      Ls[row][yv] = wv;
      Wh[row][yv] = (_Float16)wv;
    }
  }
  __syncthreads();

  // ---- attn = W @ v^T via WMMA (waves 0,1: d-tiles 0 and 16), K = 256 ----
  if (wid < 2) {
    const int nt = wid;
    v8f acc = {};
#pragma unroll
    for (int kc = 0; kc < 8; ++kc) {
      v16h a  = load_fragA(&Wh[0][kc * 32], 260, lane);
      v16h bf = load_fragP(&vB[(kc * 2 + nt) * 512 + lane * 16]);
      acc = __builtin_amdgcn_wmma_f32_16x16x32_f16(false, a, false, bf, (short)0, acc, false, false);
    }
    const int nloc = lane & 15, mh = (lane >> 4) * 8;
#pragma unroll
    for (int vi = 0; vi < 8; ++vi) Aout[nt * 16 + nloc][mh + vi] = acc[vi];
  }
  __syncthreads();

  // ---- attn_enc (Toeplitz) + final output ----
#pragma unroll
  for (int r = 0; r < 2; ++r) {
    int idx = tid + r * 256;
    int d = idx >> 4, xi = idx & 15;
    int xx = x0 + xi;
    if (xx < WLEN) {
      const float* __restrict__ rrow = rel + (32 + d) * 501 + (250 - xx);
      __builtin_prefetch(rrow, 0, 0);
      float enc = 0.f;
      for (int yv = 0; yv < WLEN; ++yv) enc += Ls[xi][yv] * rrow[yv];
      int c = h * 32 + d;
      out[((b * 256 + c) * 15 + hh) * WLEN + xx] = Aout[d][xi] + enc;
    }
  }
}

// ---------------------------------------------------------------------------
extern "C" void kernel_launch(void* const* d_in, const int* in_sizes, int n_in,
                              void* d_out, int out_size, void* d_ws, size_t ws_size,
                              hipStream_t stream) {
  (void)in_sizes; (void)n_in; (void)out_size; (void)ws_size;
  const float* x    = (const float*)d_in[0];
  const float* y    = (const float*)d_in[1];
  const float* wk   = (const float*)d_in[2];
  const float* wq   = (const float*)d_in[3];
  const float* wv   = (const float*)d_in[4];
  const float* gk   = (const float*)d_in[5];
  const float* bk   = (const float*)d_in[6];
  const float* gq   = (const float*)d_in[7];
  const float* bq   = (const float*)d_in[8];
  const float* gkq  = (const float*)d_in[9];
  const float* bkq  = (const float*)d_in[10];
  const float* glog = (const float*)d_in[11];
  const float* blog = (const float*)d_in[12];
  const float* rel  = (const float*)d_in[13];

  _Float16* Pf   = (_Float16*)d_ws;                      // 30*512*256 halves
  float* projSB  = (float*)(Pf + (size_t)NBB * 512 * WPAD);  // 512*2 scale/bias
  float* lpart   = projSB + 1024;                        // 3840*6 partials
  float* sstats  = lpart + 3840 * 6;                     // 24 scales + 8 offsets
  float* out     = (float*)d_out;

  k_proj      <<<dim3(8, 8, NBB),  256, 0, stream>>>(x, y, wk, wq, wv, Pf);
  k_projstats <<<dim3(512),        256, 0, stream>>>(Pf, gk, bk, gq, bq, gkq, bkq, projSB);
  k_logitstats<<<dim3(16, 8, NBB), 256, 0, stream>>>(Pf, projSB, rel, lpart);
  k_logitfin  <<<dim3(1),          256, 0, stream>>>(lpart, glog, blog, sstats);
  k_attn      <<<dim3(16, 8, NBB), 256, 0, stream>>>(Pf, projSB, rel, sstats, out);
}
